// UniversalTemporalGNN_8856222564558
// MI455X (gfx1250) — matrix operand order repaired and checked by reading, not compiled, not run on hardware
//
#include <hip/hip_runtime.h>
#include <hip/hip_bf16.h>
#include <math.h>

#define NN    50000
#define EE    800000
#define FF    34
#define HID   128
#define HEADS 8
#define CHN   16
#define EDIMM 8
#define BBG   64
#define LSTMH 256
#define NCLS  9

typedef __attribute__((ext_vector_type(16))) __bf16 v16bf;
typedef __attribute__((ext_vector_type(8)))  float  v8f;

__device__ __forceinline__ float sigmoidf_(float x){ return 1.f/(1.f+__expf(-x)); }

// ---------------- generic fill ----------------
__global__ void fill_kernel(float* p, float v, int n){
  int i = blockIdx.x*blockDim.x + threadIdx.x;
  if (i < n) p[i] = v;
}

// ---------------- degree + self-loop attr (fill_value='mean') ----------------
__global__ void deg_loop_kernel(const int* ei, const float* eattr, float* deg, float* lsum){
  int e = blockIdx.x*blockDim.x + threadIdx.x;
  if (e >= EE) return;
  int dst = ei[EE + e];
  atomicAdd(&deg[dst], 1.f);
  const float* ea = eattr + (size_t)e*EDIMM;
  #pragma unroll
  for (int d = 0; d < EDIMM; ++d) atomicAdd(&lsum[(size_t)dst*EDIMM + d], ea[d]);
}

__global__ void loop_fin_kernel(const float* deg, float* lattr){
  int i = blockIdx.x*blockDim.x + threadIdx.x;
  if (i >= NN*EDIMM) return;
  float d = fmaxf(deg[i / EDIMM], 1.f);
  lattr[i] = lattr[i] / d;
}

// ---------------- projection: x@W.T + b -> LN -> relu (one node per block) ----------------
__global__ void proj_kernel(const float* __restrict__ x, const float* __restrict__ W,
                            const float* __restrict__ b, const float* __restrict__ g,
                            const float* __restrict__ be, float* __restrict__ h){
  int n = blockIdx.x, j = threadIdx.x;           // blockDim = 128
  const float* xr = x + (size_t)n*FF;
  const float* wr = W + (size_t)j*FF;
  float acc = b[j];
  #pragma unroll
  for (int k = 0; k < FF; ++k) acc = fmaf(xr[k], wr[k], acc);
  __shared__ float red[HID];
  red[j] = acc; __syncthreads();
  for (int s = 64; s > 0; s >>= 1){ if (j < s) red[j] += red[j+s]; __syncthreads(); }
  float mu = red[0] * (1.f/HID); __syncthreads();
  float dv = acc - mu;
  red[j] = dv*dv; __syncthreads();
  for (int s = 64; s > 0; s >>= 1){ if (j < s) red[j] += red[j+s]; __syncthreads(); }
  float var = red[0] * (1.f/HID);
  float y = dv * rsqrtf(var + 1e-5f) * g[j] + be[j];
  h[(size_t)n*HID + j] = fmaxf(y, 0.f);
}

// ---------------- bf16 WMMA GEMM: C[M,Nout] = act(A[M,K] * W[Nout,K]^T + b1 + b2) --------
// Compile-time K (fully unrolled), one 16x64 output slab per wave32 (4 accumulators,
// A fragment converted once and reused across 4 v_wmma issues). M%16==0, Nout%64==0.
template<int K>
__global__ __launch_bounds__(128)
void wmma_gemm_kernel(const float* __restrict__ A, const float* __restrict__ W,
                      const float* __restrict__ b1, const float* __restrict__ b2,
                      float* __restrict__ C, int M, int Nout, int act){
  int wave  = threadIdx.x >> 5;
  int group = blockIdx.x * (blockDim.x >> 5) + wave;   // one wave = one 16x64 slab
  int ngn   = Nout >> 6;
  int total = (M >> 4) * ngn;
  if (group >= total) return;                          // whole-wave exit: EXEC all-1s for WMMA
  int mt = group / ngn, ng = group % ngn;
  int lane = threadIdx.x & 31;
  int l16 = lane & 15, half = lane >> 4;
  const float* arow  = A + (size_t)(mt*16 + l16) * K;      // A row (M = lane%16)
  const float* wbase = W + (size_t)(ng*64 + l16) * K;      // weight rows == B columns
  v8f acc[4] = {};
  #pragma unroll
  for (int k0 = 0; k0 < K; k0 += 32){
    v16bf a;
    #pragma unroll
    for (int v = 0; v < 8; ++v){
      int ka = k0 + ((v < 4) ? 0 : 16) + half*8 + (v & 3)*2;  // 16-bit A 16x32 layout (§7.12.2)
      float2 f = *(const float2*)(arow + ka);
      a[2*v]   = (__bf16)f.x;
      a[2*v+1] = (__bf16)f.y;
    }
    #pragma unroll
    for (int t = 0; t < 4; ++t){
      const float* wrow = wbase + (size_t)t * 16 * K;
      v16bf bm;
      #pragma unroll
      for (int v = 0; v < 8; ++v){
        int kb = k0 + ((v < 4) ? 0 : 16) + half*8 + (v & 3)*2;  // 16-bit B 32x16 layout
        float2 f = *(const float2*)(wrow + kb);
        bm[2*v]   = (__bf16)f.x;
        bm[2*v+1] = (__bf16)f.y;
      }
      acc[t] = __builtin_amdgcn_wmma_f32_16x16x32_bf16(false, a, false, bm,
                                                       (short)0, acc[t], false, false);
    }
  }
  #pragma unroll
  for (int t = 0; t < 4; ++t){
    int n = ng*64 + t*16 + l16;
    float bias = (b1 ? b1[n] : 0.f) + (b2 ? b2[n] : 0.f);
    #pragma unroll
    for (int v = 0; v < 8; ++v){
      int m = mt*16 + v + half*8;                   // C/D layout: VGPR v -> M = v (+8 hi half)
      float val = acc[t][v] + bias;
      if (act == 1) val = fmaxf(val, 0.f);
      C[(size_t)m*Nout + n] = val;
    }
  }
}

// ---------------- GATv2 edge pass 1: logits + segment max ----------------
__global__ void edge_logit_kernel(const int* __restrict__ ei, const float* __restrict__ eattr,
                                  const float* __restrict__ lattr,
                                  const float* __restrict__ xl, const float* __restrict__ xr,
                                  const float* __restrict__ We, const float* __restrict__ att,
                                  float* __restrict__ elog, float* __restrict__ mx){
  int idx = blockIdx.x*blockDim.x + threadIdx.x;
  if (idx >= (EE+NN)*HEADS) return;
  int e = idx >> 3, hh = idx & 7;
  int src, dst; const float* ea;
  if (e < EE){ src = ei[e]; dst = ei[EE+e]; ea = eattr + (size_t)e*EDIMM; }
  else       { src = dst = e - EE;          ea = lattr + (size_t)(e-EE)*EDIMM; }
  const float* xls = xl + (size_t)src*HID + hh*CHN;
  const float* xrd = xr + (size_t)dst*HID + hh*CHN;
  const float* atv = att + hh*CHN;
  float logit = 0.f;
  #pragma unroll
  for (int c = 0; c < CHN; ++c){
    const float* wr = We + (size_t)(hh*CHN + c)*EDIMM;
    float ev = 0.f;
    #pragma unroll
    for (int d = 0; d < EDIMM; ++d) ev = fmaf(ea[d], wr[d], ev);
    float m = xls[c] + xrd[c] + ev;
    m = (m > 0.f) ? m : 0.2f*m;                     // leaky_relu(0.2)
    logit = fmaf(m, atv[c], logit);
  }
  elog[idx] = logit;
  float* addr = &mx[(size_t)dst*HEADS + hh];        // float atomic-max via int/uint trick
  if (logit >= 0.f) atomicMax((int*)addr, __float_as_int(logit));
  else              atomicMin((unsigned int*)addr, __float_as_uint(logit));
}

// ---------------- GATv2 edge pass 2: exp + segment sum ----------------
__global__ void edge_exp_kernel(const int* __restrict__ ei, const float* __restrict__ mx,
                                float* __restrict__ elog, float* __restrict__ den){
  int idx = blockIdx.x*blockDim.x + threadIdx.x;
  if (idx >= (EE+NN)*HEADS) return;
  int e = idx >> 3, hh = idx & 7;
  int dst = (e < EE) ? ei[EE+e] : (e - EE);
  float ex = __expf(elog[idx] - mx[(size_t)dst*HEADS + hh]);
  elog[idx] = ex;
  atomicAdd(&den[(size_t)dst*HEADS + hh], ex);
}

// ---------------- GATv2 edge pass 3: weighted scatter-add of xl[src] ----------------
__global__ void edge_aggr_kernel(const int* __restrict__ ei, const float* __restrict__ elog,
                                 const float* __restrict__ den, const float* __restrict__ xl,
                                 float* __restrict__ aout){
  int idx = blockIdx.x*blockDim.x + threadIdx.x;
  if (idx >= (EE+NN)*HID) return;
  int e = idx >> 7, c = idx & 127, hh = c >> 4;
  int src, dst;
  if (e < EE){ src = ei[e]; dst = ei[EE+e]; } else { src = dst = e - EE; }
  float alpha = elog[(size_t)e*HEADS + hh] / den[(size_t)dst*HEADS + hh];
  atomicAdd(&aout[(size_t)dst*HID + c], alpha * xl[(size_t)src*HID + c]);
}

// ---------------- GAT epilogue: +bias, LN, optional residual, relu ----------------
__global__ void gat_post_kernel(const float* __restrict__ aout, const float* __restrict__ bias,
                                const float* __restrict__ lg, const float* __restrict__ lb,
                                float* __restrict__ h, int residual){
  int n = blockIdx.x, j = threadIdx.x;             // blockDim = 128
  float acc = aout[(size_t)n*HID + j] + bias[j];
  __shared__ float red[HID];
  red[j] = acc; __syncthreads();
  for (int s = 64; s > 0; s >>= 1){ if (j < s) red[j] += red[j+s]; __syncthreads(); }
  float mu = red[0] * (1.f/HID); __syncthreads();
  float dv = acc - mu;
  red[j] = dv*dv; __syncthreads();
  for (int s = 64; s > 0; s >>= 1){ if (j < s) red[j] += red[j+s]; __syncthreads(); }
  float var = red[0] * (1.f/HID);
  float y = dv * rsqrtf(var + 1e-5f) * lg[j] + lb[j];
  float hv = residual ? (h[(size_t)n*HID + j] + y) : y;
  h[(size_t)n*HID + j] = fmaxf(hv, 0.f);
}

// ---------------- generic second-layer head: out = act(in@W2.T + b2) ----------------
__global__ void head2_kernel(const float* __restrict__ in, const float* __restrict__ W2,
                             const float* __restrict__ b2, float* __restrict__ out,
                             int M, int O, int K, int act){
  int idx = blockIdx.x*blockDim.x + threadIdx.x;
  if (idx >= M*O) return;
  int m = idx / O, o = idx % O;
  const float* ir = in + (size_t)m*K;
  const float* wr = W2 + (size_t)o*K;
  float acc = b2[o];
  for (int k = 0; k < K; ++k) acc = fmaf(ir[k], wr[k], acc);
  if (act == 1) acc = sigmoidf_(acc);
  else if (act == 2) acc = (acc > 20.f) ? acc : log1pf(__expf(acc));  // softplus
  out[idx] = acc;
}

// ---------------- mean pooling over batch ----------------
__global__ void pool_kernel(const float* __restrict__ h, const int* __restrict__ batch,
                            float* __restrict__ pooled, float* __restrict__ cnt){
  int idx = blockIdx.x*blockDim.x + threadIdx.x;
  if (idx >= NN*HID) return;
  int n = idx >> 7, j = idx & 127;
  int b = batch[n];
  atomicAdd(&pooled[(size_t)b*HID + j], h[idx]);
  if (j == 0) atomicAdd(&cnt[b], 1.f);
}

__global__ void pool_div_kernel(float* pooled, const float* cnt){
  int idx = blockIdx.x*blockDim.x + threadIdx.x;
  if (idx >= BBG*HID) return;
  pooled[idx] /= fmaxf(cnt[idx >> 7], 1.f);
}

// ---------------- LSTM gate nonlinearity (zero initial h,c; torch order i,f,g,o) --------
__global__ void lstm_gate_kernel(const float* __restrict__ gates, float* __restrict__ hout){
  int idx = blockIdx.x*blockDim.x + threadIdx.x;
  if (idx >= BBG*LSTMH) return;
  int b = idx / LSTMH, j = idx % LSTMH;
  const float* g = gates + (size_t)b * 4 * LSTMH;
  float i = g[j], gg = g[2*LSTMH + j], o = g[3*LSTMH + j];
  float c = sigmoidf_(i) * tanhf(gg);
  hout[idx] = sigmoidf_(o) * tanhf(c);
}

// =====================================================================================
extern "C" void kernel_launch(void* const* d_in, const int* in_sizes, int n_in,
                              void* d_out, int out_size, void* d_ws, size_t ws_size,
                              hipStream_t stream) {
  (void)in_sizes; (void)n_in; (void)out_size; (void)ws_size;
  // Input order: recursive insertion-order flattening of setup_inputs():
  // 0:x 1:edge_index 2:edge_attr 3:batch, then params:
  // 4..7: proj_W proj_b proj_g proj_be
  // 8+9i: gat_i {Wl,bl,Wr,br,We,att,bias,ln_g,ln_b}
  // 35..38: ch_W1 ch_b1 ch_W2 ch_b2   39..42: ca_W1 ca_b1 ca_W2 ca_b2
  // 43..46: lstm0 Wih Whh bih bhh     47..50: lstm1 Wih Whh bih bhh
  // 51..54: gh  55..58: gd  59..62: ga  63..66: gr  (each W1 b1 W2 b2)
  auto F = [&](int i){ return (const float*)d_in[i]; };
  const float* x     = F(0);
  const int*   ei    = (const int*)d_in[1];
  const float* eattr = F(2);
  const int*   batch = (const int*)d_in[3];

  float* ws = (float*)d_ws;
  size_t o = 0;
  float* h     = ws + o; o += (size_t)NN*HID;
  float* xl    = ws + o; o += (size_t)NN*HID;     // reused as ch hidden t1 later
  float* xr    = ws + o; o += (size_t)NN*HID;     // reused as ca hidden t2 later
  float* aout  = ws + o; o += (size_t)NN*HID;
  float* elog  = ws + o; o += (size_t)(EE+NN)*HEADS;
  float* mx    = ws + o; o += (size_t)NN*HEADS;
  float* den   = ws + o; o += (size_t)NN*HEADS;
  float* deg   = ws + o; o += (size_t)NN;
  float* lattr = ws + o; o += (size_t)NN*EDIMM;
  float* pooled= ws + o; o += (size_t)BBG*HID;
  float* cnt   = ws + o; o += (size_t)BBG;
  float* gates = ws + o; o += (size_t)BBG*4*LSTMH;
  float* h1    = ws + o; o += (size_t)BBG*LSTMH;
  float* lstm  = ws + o; o += (size_t)BBG*LSTMH;
  float* thead = ws + o; o += (size_t)BBG*HID;

  float* out = (float*)d_out;
  float* out_ch = out;                          // [N,1]
  float* out_ca = out + NN;                     // [N,9]
  float* out_gh = out + NN + NN*NCLS;           // [B,1]
  float* out_gd = out_gh + BBG;                 // [B,1]
  float* out_ga = out_gd + BBG;                 // [B,9]
  float* out_gr = out_ga + BBG*NCLS;            // [B,1]

  const int T = 256;
  auto grid = [&](long long n){ return dim3((unsigned)((n + T - 1) / T)); };
  auto gemm = [&](const float* A, const float* W, const float* b1, const float* b2,
                  float* C, int M, int Nout, int K, int act){
    int waves  = (M/16) * (Nout/64);              // one wave per 16x64 output slab
    int blocks = (waves + 3) / 4;                 // 128 threads = 4 waves per block
    if (K == 128)
      wmma_gemm_kernel<128><<<blocks, 128, 0, stream>>>(A, W, b1, b2, C, M, Nout, act);
    else
      wmma_gemm_kernel<256><<<blocks, 128, 0, stream>>>(A, W, b1, b2, C, M, Nout, act);
  };

  // ---- self-loop edge attr (mean of incoming) ----
  fill_kernel<<<grid(NN), T, 0, stream>>>(deg, 0.f, NN);
  fill_kernel<<<grid(NN*EDIMM), T, 0, stream>>>(lattr, 0.f, NN*EDIMM);
  deg_loop_kernel<<<grid(EE), T, 0, stream>>>(ei, eattr, deg, lattr);
  loop_fin_kernel<<<grid(NN*EDIMM), T, 0, stream>>>(deg, lattr);

  // ---- projection + LN + relu ----
  proj_kernel<<<NN, HID, 0, stream>>>(x, F(4), F(5), F(6), F(7), h);

  // ---- 3 GATv2 layers ----
  for (int layer = 0; layer < 3; ++layer){
    int gi = 8 + layer*9;
    const float *Wl = F(gi+0), *bl = F(gi+1), *Wr = F(gi+2), *br = F(gi+3);
    const float *We = F(gi+4), *att = F(gi+5), *gbias = F(gi+6);
    const float *lng = F(gi+7), *lnb = F(gi+8);
    gemm(h, Wl, bl, nullptr, xl, NN, HID, HID, 0);
    gemm(h, Wr, br, nullptr, xr, NN, HID, HID, 0);
    fill_kernel<<<grid(NN*HEADS), T, 0, stream>>>(mx, -1e30f, NN*HEADS);
    fill_kernel<<<grid(NN*HEADS), T, 0, stream>>>(den, 0.f, NN*HEADS);
    fill_kernel<<<grid((long long)NN*HID), T, 0, stream>>>(aout, 0.f, NN*HID);
    edge_logit_kernel<<<grid((long long)(EE+NN)*HEADS), T, 0, stream>>>(
        ei, eattr, lattr, xl, xr, We, att, elog, mx);
    edge_exp_kernel<<<grid((long long)(EE+NN)*HEADS), T, 0, stream>>>(ei, mx, elog, den);
    edge_aggr_kernel<<<grid((long long)(EE+NN)*HID), T, 0, stream>>>(ei, elog, den, xl, aout);
    gat_post_kernel<<<NN, HID, 0, stream>>>(aout, gbias, lng, lnb, h, layer > 0 ? 1 : 0);
  }

  // ---- node heads ----
  gemm(h, F(35), F(36), nullptr, xl, NN, 64, HID, 1);           // ch hidden (relu)
  head2_kernel<<<grid(NN), T, 0, stream>>>(xl, F(37), F(38), out_ch, NN, 1, 64, 1);
  gemm(h, F(39), F(40), nullptr, xr, NN, HID, HID, 1);          // ca hidden (relu)
  head2_kernel<<<grid((long long)NN*NCLS), T, 0, stream>>>(xr, F(41), F(42), out_ca, NN, NCLS, HID, 0);

  // ---- mean pool per graph ----
  fill_kernel<<<grid(BBG*HID), T, 0, stream>>>(pooled, 0.f, BBG*HID);
  fill_kernel<<<1, T, 0, stream>>>(cnt, 0.f, BBG);
  pool_kernel<<<grid((long long)NN*HID), T, 0, stream>>>(h, batch, pooled, cnt);
  pool_div_kernel<<<grid(BBG*HID), T, 0, stream>>>(pooled, cnt);

  // ---- 2-layer LSTM single step (zero init state) ----
  gemm(pooled, F(43), F(45), F(46), gates, BBG, 4*LSTMH, HID, 0);
  lstm_gate_kernel<<<grid(BBG*LSTMH), T, 0, stream>>>(gates, h1);
  gemm(h1, F(47), F(49), F(50), gates, BBG, 4*LSTMH, LSTMH, 0);
  lstm_gate_kernel<<<grid(BBG*LSTMH), T, 0, stream>>>(gates, lstm);

  // ---- graph heads ----
  gemm(lstm, F(51), F(52), nullptr, thead, BBG, 64, LSTMH, 1);
  head2_kernel<<<1, T, 0, stream>>>(thead, F(53), F(54), out_gh, BBG, 1, 64, 1);
  gemm(lstm, F(55), F(56), nullptr, thead, BBG, 64, LSTMH, 1);
  head2_kernel<<<1, T, 0, stream>>>(thead, F(57), F(58), out_gd, BBG, 1, 64, 1);
  gemm(lstm, F(59), F(60), nullptr, thead, BBG, 128, LSTMH, 1);
  head2_kernel<<<grid(BBG*NCLS), T, 0, stream>>>(thead, F(61), F(62), out_ga, BBG, NCLS, 128, 0);
  gemm(lstm, F(63), F(64), nullptr, thead, BBG, 128, LSTMH, 1);
  head2_kernel<<<1, T, 0, stream>>>(thead, F(65), F(66), out_gr, BBG, 1, 128, 2);
}
